// TransMIL_end_22488448762258
// MI455X (gfx1250) — compile-verified, matrix operand order checked by reading, not run on hardware
//
#include <hip/hip_runtime.h>

typedef _Float16 half_t;
typedef __attribute__((ext_vector_type(16))) _Float16 v16h;
typedef __attribute__((ext_vector_type(8)))  _Float16 h8;
typedef __attribute__((ext_vector_type(8)))  float    v8f;

// ---------------- problem constants ----------------
constexpr int BATCH = 2;
constexpr int NIN   = 8192;          // raw tokens
constexpr int OUTER = 1024;
constexpr int DIM   = 512;
constexpr int HEADS = 8;
constexpr int DH    = 64;
constexpr int HS    = 91;            // ceil(sqrt(8192))
constexpr int NSEQ  = 1 + HS * HS;   // 8282 (cls + 8281)
constexpr int NTOK  = 8448;          // padded to 33*256
constexpr int PAD   = NTOK - NSEQ;   // 166 (front padding)
constexpr int LAND  = 256;
constexpr int NL    = NTOK / LAND;   // 33
constexpr int RESK  = 33;
constexpr int BH    = BATCH * HEADS; // 16

// ---------------- WMMA GEMM ----------------
// C[M,N] = alpha * A[M,K] x Op(B) (+bias) (+relu), fp32 accumulate.
// TB==0: B is (N,K) row-major  -> C = A @ B^T   ("x @ W.T" form)
// TB==1: B is (K,N) row-major  -> C = A @ B     (staged transposed into LDS)
// Batched via blockIdx.z with element strides sA/sB/sC. C and/or Ch may be null.
//
// ALIGNMENT CONTRACT (holds for every GEMM in this network, so no guards):
//   M % 128 == 0, N % 64 == 0, K % 64 == 0, lda/ldb/ldc % 8 == 0.
#define BM 128
#define BN 64
#define BK 64
#define ASTR 72   // 64 + 8 halfs pad; row stride 144B (16B aligned)
#define BSTR 72

template<int TB>
__global__ __launch_bounds__(256)
void gemm16(const half_t* __restrict__ A, const half_t* __restrict__ Bm,
            float* __restrict__ C, half_t* __restrict__ Ch,
            const float* __restrict__ bias,
            int M, int N, int K, int lda, int ldb, int ldc,
            long long sA, long long sB, long long sC,
            float alpha, int relu)
{
    __shared__ __align__(16) half_t As[BM * ASTR];
    __shared__ __align__(16) half_t Bs[BN * BSTR];

    const int tid = threadIdx.x;
    A  += (long long)blockIdx.z * sA;
    Bm += (long long)blockIdx.z * sB;
    const long long cOff = (long long)blockIdx.z * sC;

    const int rowBase = blockIdx.y * BM;
    const int colBase = blockIdx.x * BN;
    const int wave = tid >> 5, lane = tid & 31;
    const int lm = lane & 15, hi = lane >> 4;
    const int kb = hi * 8;          // K sub-pattern base per ISA 16-bit A/B layout
    const int mrow = wave * 16;

    // staging coordinates (straight-line, no bounds checks)
    const int ar  = tid >> 1;             // A row within tile (0..127)
    const int akk = (tid & 1) << 5;       // A k-offset: 0 or 32
    const int bn  = tid >> 2;             // TB0: n within tile (0..63)
    const int bkk = (tid & 3) << 4;       // TB0: k-offset 0/16/32/48
    const int tk  = tid >> 2;             // TB1: k within tile (0..63)
    const int tnn = (tid & 3) << 4;       // TB1: n-offset 0/16/32/48

    v8f zero8 = {0.f, 0.f, 0.f, 0.f, 0.f, 0.f, 0.f, 0.f};
    v8f acc[4];
#pragma unroll
    for (int i = 0; i < 4; ++i) acc[i] = zero8;

    const half_t* aSrc = A + (long long)(rowBase + ar) * lda + akk;
    const half_t* bSrc = (TB == 0)
        ? Bm + (long long)(colBase + bn) * ldb + bkk
        : Bm + (long long)tk * ldb + colBase + tnn;
    const long long bStep = (TB == 0) ? (long long)BK : (long long)BK * ldb;

    for (int k0 = 0; k0 < K; k0 += BK) {
        // ---- stage A tile: 128 rows x 64 halfs (32 halfs / thread) ----
        {
            half_t* dst = &As[ar * ASTR + akk];
            *(h8*)(dst)      = *(const h8*)(aSrc);
            *(h8*)(dst + 8)  = *(const h8*)(aSrc + 8);
            *(h8*)(dst + 16) = *(const h8*)(aSrc + 16);
            *(h8*)(dst + 24) = *(const h8*)(aSrc + 24);
            __builtin_prefetch((const void*)(aSrc + BK), 0, 1);
            aSrc += BK;
        }
        // ---- stage B tile: 64 n-rows x 64 halfs ----
        if (TB == 0) {
            half_t* dst = &Bs[bn * BSTR + bkk];
            *(h8*)(dst)     = *(const h8*)(bSrc);
            *(h8*)(dst + 8) = *(const h8*)(bSrc + 8);
            __builtin_prefetch((const void*)(bSrc + BK), 0, 1);
        } else {
            h8 v0 = *(const h8*)(bSrc);
            h8 v1 = *(const h8*)(bSrc + 8);
            __builtin_prefetch((const void*)(bSrc + bStep), 0, 1);
#pragma unroll
            for (int j = 0; j < 8; ++j) {
                Bs[(tnn + j) * BSTR + tk]     = v0[j];
                Bs[(tnn + 8 + j) * BSTR + tk] = v1[j];
            }
        }
        bSrc += bStep;
        __syncthreads();

        // ---- 8 WMMAs per staging round: 2 K-steps x 4 N-subtiles ----
        union Fr { v16h v; h8 h[2]; };
#pragma unroll
        for (int ks = 0; ks < BK; ks += 32) {
            Fr a;
            a.h[0] = *(const h8*)&As[(mrow + lm) * ASTR + ks + kb];
            a.h[1] = *(const h8*)&As[(mrow + lm) * ASTR + ks + kb + 16];
#pragma unroll
            for (int nt = 0; nt < 4; ++nt) {
                Fr b;
                b.h[0] = *(const h8*)&Bs[(nt * 16 + lm) * BSTR + ks + kb];
                b.h[1] = *(const h8*)&Bs[(nt * 16 + lm) * BSTR + ks + kb + 16];
                acc[nt] = __builtin_amdgcn_wmma_f32_16x16x32_f16(
                    false, a.v, false, b.v, (short)0, acc[nt], false, false);
            }
        }
        __syncthreads();
    }

    // ---- epilogue: C layout VGPR j -> M=j+8*hi, N=lane&15 (no guards) ----
#pragma unroll
    for (int nt = 0; nt < 4; ++nt) {
        int gn = colBase + nt * 16 + lm;
        float bs = bias ? bias[gn] : 0.f;
#pragma unroll
        for (int j = 0; j < 8; ++j) {
            int gm = rowBase + mrow + hi * 8 + j;
            float v = acc[nt][j] * alpha + bs;
            if (relu) v = v > 0.f ? v : 0.f;
            long long idx = cOff + (long long)gm * ldc + gn;
            if (C)  C[idx]  = v;
            if (Ch) Ch[idx] = (half_t)v;
        }
    }
}

// ---------------- elementwise / reduction kernels ----------------
__global__ void k_cast_f16(const float* __restrict__ x, half_t* __restrict__ y, size_t n) {
    size_t i = (size_t)blockIdx.x * 256 + threadIdx.x;
    if (i < n) y[i] = (half_t)x[i];
}
__global__ void k_zero_f16(half_t* p, size_t n) {
    size_t i = (size_t)blockIdx.x * 256 + threadIdx.x;
    if (i < n) p[i] = (half_t)0.f;
}
__global__ void k_zero_f32(float* p, size_t n) {
    size_t i = (size_t)blockIdx.x * 256 + threadIdx.x;
    if (i < n) p[i] = 0.f;
}

// h = [cls, H0[0..8191], H0[0..88]]  (B, NSEQ, DIM)
__global__ void k_build_h(const float* __restrict__ H0, const float* __restrict__ cls,
                          float* __restrict__ h) {
    size_t idx = (size_t)blockIdx.x * 256 + threadIdx.x;
    size_t total = (size_t)BATCH * NSEQ * DIM;
    if (idx >= total) return;
    int c = idx % DIM;
    size_t t = idx / DIM;
    int i = t % NSEQ;
    int b = t / NSEQ;
    float v;
    if (i == 0) v = cls[c];
    else {
        int r = i - 1;
        if (r >= NIN) r -= NIN;
        v = H0[((size_t)b * NIN + r) * DIM + c];
    }
    h[idx] = v;
}

// LayerNorm over DIM, written f16 into front-padded token buffer (B, NTOK, DIM)
__global__ void k_ln_pad(const float* __restrict__ X, const float* __restrict__ g,
                         const float* __restrict__ bta, half_t* __restrict__ Y) {
    int row = blockIdx.x;                 // 0 .. B*NSEQ-1
    int b = row / NSEQ, i = row % NSEQ;
    const float* x = X + (size_t)row * DIM;
    __shared__ float red[256];
    int tid = threadIdx.x;
    float s = 0.f;
    for (int c = tid; c < DIM; c += 256) s += x[c];
    red[tid] = s; __syncthreads();
    for (int o = 128; o > 0; o >>= 1) { if (tid < o) red[tid] += red[tid + o]; __syncthreads(); }
    float mu = red[0] / DIM; __syncthreads();
    float v = 0.f;
    for (int c = tid; c < DIM; c += 256) { float d = x[c] - mu; v += d * d; }
    red[tid] = v; __syncthreads();
    for (int o = 128; o > 0; o >>= 1) { if (tid < o) red[tid] += red[tid + o]; __syncthreads(); }
    float rstd = rsqrtf(red[0] / DIM + 1e-5f);
    half_t* y = Y + ((size_t)b * NTOK + PAD + i) * DIM;
    for (int c = tid; c < DIM; c += 256)
        y[c] = (half_t)((x[c] - mu) * rstd * g[c] + bta[c]);
}

// split QKV (B,NTOK,1536) -> Q,K,V (B,H,NTOK,DH); q scaled by DH^-0.5
__global__ void k_split_qkv(const float* __restrict__ QKV, half_t* __restrict__ Q,
                            half_t* __restrict__ Kh, half_t* __restrict__ V,
                            float* __restrict__ Vf) {
    size_t idx = (size_t)blockIdx.x * 256 + threadIdx.x;
    size_t total = (size_t)BH * NTOK * DH;
    if (idx >= total) return;
    int d = idx % DH;
    size_t t = idx / DH;
    int i = t % NTOK; t /= NTOK;
    int h = t % HEADS;
    int b = t / HEADS;
    size_t row = ((size_t)b * NTOK + i) * (3 * DIM);
    float q = QKV[row + h * DH + d] * 0.125f;
    float k = QKV[row + DIM + h * DH + d];
    float v = QKV[row + 2 * DIM + h * DH + d];
    Q[idx] = (half_t)q; Kh[idx] = (half_t)k; V[idx] = (half_t)v; Vf[idx] = v;
}

// landmarks: XL[bh,m,d] = mean_j X[bh, m*NL+j, d]
__global__ void k_landmarks(const half_t* __restrict__ X, half_t* __restrict__ XL) {
    size_t idx = (size_t)blockIdx.x * 256 + threadIdx.x;
    size_t total = (size_t)BH * LAND * DH;
    if (idx >= total) return;
    int d = idx % DH;
    size_t t = idx / DH;
    int m = t % LAND;
    int bh = t / LAND;
    const half_t* src = X + ((size_t)bh * NTOK + (size_t)m * NL) * DH + d;
    float s = 0.f;
    for (int j = 0; j < NL; ++j) s += (float)src[(size_t)j * DH];
    XL[idx] = (half_t)(s / NL);
}

// row softmax (f32 in, f16 out)
__global__ void k_softmax(const float* __restrict__ X, half_t* __restrict__ Y, int cols) {
    size_t row = blockIdx.x;
    const float* x = X + row * (size_t)cols;
    half_t* y = Y + row * (size_t)cols;
    __shared__ float red[256];
    int tid = threadIdx.x;
    float m = -3.4e38f;
    for (int c = tid; c < cols; c += 256) m = fmaxf(m, x[c]);
    red[tid] = m; __syncthreads();
    for (int o = 128; o > 0; o >>= 1) { if (tid < o) red[tid] = fmaxf(red[tid], red[tid + o]); __syncthreads(); }
    m = red[0]; __syncthreads();
    float s = 0.f;
    for (int c = tid; c < cols; c += 256) s += __expf(x[c] - m);
    red[tid] = s; __syncthreads();
    for (int o = 128; o > 0; o >>= 1) { if (tid < o) red[tid] += red[tid + o]; __syncthreads(); }
    float inv = 1.f / red[0];
    for (int c = tid; c < cols; c += 256) y[c] = (half_t)(__expf(x[c] - m) * inv);
}

// global max of |row sums| / |col sums| of a2 (non-negative -> uint-bit atomicMax ok)
__global__ void k_a2_rowmax(const half_t* __restrict__ A2, float* scal) {
    int r = blockIdx.x * 256 + threadIdx.x;
    if (r >= BH * LAND) return;
    const half_t* row = A2 + (size_t)r * LAND;
    float s = 0.f;
    for (int j = 0; j < LAND; ++j) s += fabsf((float)row[j]);
    atomicMax((unsigned int*)&scal[0], __float_as_uint(s));
}
__global__ void k_a2_colmax(const half_t* __restrict__ A2, float* scal) {
    int ci = blockIdx.x * 256 + threadIdx.x;
    if (ci >= BH * LAND) return;
    int col = ci % LAND, bh = ci / LAND;
    const half_t* p = A2 + (size_t)bh * LAND * LAND + col;
    float s = 0.f;
    for (int r = 0; r < LAND; ++r) s += fabsf((float)p[(size_t)r * LAND]);
    atomicMax((unsigned int*)&scal[1], __float_as_uint(s));
}
// Z0 = a2^T / (maxRowSum * maxColSum)
__global__ void k_pinv_seed(const half_t* __restrict__ A2, half_t* __restrict__ Z,
                            const float* __restrict__ scal) {
    size_t idx = (size_t)blockIdx.x * 256 + threadIdx.x;
    size_t total = (size_t)BH * LAND * LAND;
    if (idx >= total) return;
    int j = idx % LAND;
    size_t t = idx / LAND;
    int i = t % LAND;
    int bh = t / LAND;
    float inv = 1.f / (scal[0] * scal[1]);
    Z[idx] = (half_t)((float)A2[((size_t)bh * LAND + j) * LAND + i] * inv);
}
// Y = c*I - X   (f32 in, f16 out)
__global__ void k_ci_minus(float cst, const float* __restrict__ X, half_t* __restrict__ Y) {
    size_t idx = (size_t)blockIdx.x * 256 + threadIdx.x;
    size_t total = (size_t)BH * LAND * LAND;
    if (idx >= total) return;
    int j = idx % LAND;
    int i = (idx / LAND) % LAND;
    Y[idx] = (half_t)(((i == j) ? cst : 0.f) - X[idx]);
}

// depthwise 33-tap conv over tokens, added into OUT (B,H,NTOK,DH)
__global__ void k_conv_res_add(const float* __restrict__ Vf, const float* __restrict__ w,
                               float* __restrict__ OUT) {
    size_t idx = (size_t)blockIdx.x * 256 + threadIdx.x;
    size_t total = (size_t)BH * NTOK * DH;
    if (idx >= total) return;
    int d = idx % DH;
    size_t t = idx / DH;
    int i = t % NTOK;
    int bh = t / NTOK;
    int h = bh % HEADS;
    const float* vb = Vf + (size_t)bh * NTOK * DH + d;
    float s = 0.f;
    for (int tt = 0; tt < RESK; ++tt) {
        int ii = i + tt - RESK / 2;
        if (ii >= 0 && ii < NTOK) s += vb[(size_t)ii * DH] * w[h * RESK + tt];
    }
    OUT[idx] += s;
}

// (B,H,NTOK,DH) -> (B,NTOK,H*DH) as f16
__global__ void k_merge_heads(const float* __restrict__ OUT, half_t* __restrict__ Y) {
    size_t idx = (size_t)blockIdx.x * 256 + threadIdx.x;
    size_t total = (size_t)BATCH * NTOK * DIM;
    if (idx >= total) return;
    int c = idx % DIM;
    size_t t = idx / DIM;
    int i = t % NTOK;
    int b = t / NTOK;
    int h = c / DH, d = c % DH;
    Y[idx] = (half_t)OUT[(((size_t)(b * HEADS + h) * NTOK) + i) * DH + d];
}

// h += proj[:, PAD:]   (drop front padding)
__global__ void k_add_res(float* __restrict__ h, const float* __restrict__ proj) {
    size_t idx = (size_t)blockIdx.x * 256 + threadIdx.x;
    size_t total = (size_t)BATCH * NSEQ * DIM;
    if (idx >= total) return;
    int c = idx % DIM;
    size_t t = idx / DIM;
    int i = t % NSEQ;
    int b = t / NSEQ;
    h[idx] += proj[((size_t)b * NTOK + PAD + i) * DIM + c];
}

// PPEG: hout[1+s] = feat + dw7 + dw5 + dw3 (+biases); kernels cross-correlate
__global__ void k_ppeg(const float* __restrict__ hin, float* __restrict__ hout,
                       const float* __restrict__ w7, const float* __restrict__ b7,
                       const float* __restrict__ w5, const float* __restrict__ b5,
                       const float* __restrict__ w3, const float* __restrict__ b3) {
    size_t idx = (size_t)blockIdx.x * 256 + threadIdx.x;
    size_t total = (size_t)BATCH * HS * HS * DIM;
    if (idx >= total) return;
    int c = idx % DIM;
    size_t t = idx / DIM;
    int s = t % (HS * HS);
    int b = t / (HS * HS);
    int y = s / HS, x = s % HS;
    const float* base = hin + ((size_t)b * NSEQ + 1) * DIM;  // feature rows
    float acc = base[(size_t)s * DIM + c] + b7[c] + b5[c] + b3[c];
    for (int ky = 0; ky < 7; ++ky) {
        int yy = y + ky - 3; if (yy < 0 || yy >= HS) continue;
        for (int kx = 0; kx < 7; ++kx) {
            int xx = x + kx - 3; if (xx < 0 || xx >= HS) continue;
            acc += w7[c * 49 + ky * 7 + kx] * base[((size_t)yy * HS + xx) * DIM + c];
        }
    }
    for (int ky = 0; ky < 5; ++ky) {
        int yy = y + ky - 2; if (yy < 0 || yy >= HS) continue;
        for (int kx = 0; kx < 5; ++kx) {
            int xx = x + kx - 2; if (xx < 0 || xx >= HS) continue;
            acc += w5[c * 25 + ky * 5 + kx] * base[((size_t)yy * HS + xx) * DIM + c];
        }
    }
    for (int ky = 0; ky < 3; ++ky) {
        int yy = y + ky - 1; if (yy < 0 || yy >= HS) continue;
        for (int kx = 0; kx < 3; ++kx) {
            int xx = x + kx - 1; if (xx < 0 || xx >= HS) continue;
            acc += w3[c * 9 + ky * 3 + kx] * base[((size_t)yy * HS + xx) * DIM + c];
        }
    }
    hout[((size_t)b * NSEQ + 1 + s) * DIM + c] = acc;
}
__global__ void k_copy_cls(const float* __restrict__ hin, float* __restrict__ hout) {
    int i = blockIdx.x * 256 + threadIdx.x;
    if (i >= BATCH * DIM) return;
    int b = i / DIM, c = i % DIM;
    hout[(size_t)b * NSEQ * DIM + c] = hin[(size_t)b * NSEQ * DIM + c];
}

// final LN on token 0 + 2-class head
__global__ void k_final(const float* __restrict__ h, const float* __restrict__ g,
                        const float* __restrict__ bta, const float* __restrict__ fcw,
                        const float* __restrict__ fcb, float* __restrict__ out) {
    int b = blockIdx.x;
    const float* x = h + (size_t)b * NSEQ * DIM;
    __shared__ float red[256];
    __shared__ float lnv[DIM];
    int tid = threadIdx.x;
    float s = 0.f;
    for (int c = tid; c < DIM; c += 256) s += x[c];
    red[tid] = s; __syncthreads();
    for (int o = 128; o > 0; o >>= 1) { if (tid < o) red[tid] += red[tid + o]; __syncthreads(); }
    float mu = red[0] / DIM; __syncthreads();
    float v = 0.f;
    for (int c = tid; c < DIM; c += 256) { float d = x[c] - mu; v += d * d; }
    red[tid] = v; __syncthreads();
    for (int o = 128; o > 0; o >>= 1) { if (tid < o) red[tid] += red[tid + o]; __syncthreads(); }
    float rstd = rsqrtf(red[0] / DIM + 1e-5f); __syncthreads();
    for (int c = tid; c < DIM; c += 256) lnv[c] = (x[c] - mu) * rstd * g[c] + bta[c];
    __syncthreads();
    for (int o = 0; o < 2; ++o) {
        float d = 0.f;
        for (int c = tid; c < DIM; c += 256) d += lnv[c] * fcw[o * DIM + c];
        red[tid] = d; __syncthreads();
        for (int q = 128; q > 0; q >>= 1) { if (tid < q) red[tid] += red[tid + q]; __syncthreads(); }
        if (tid == 0) out[b * 2 + o] = red[0] + fcb[o];
        __syncthreads();
    }
}

// ---------------- host helpers ----------------
static inline dim3 ew_grid(size_t n) { return dim3((unsigned)((n + 255) / 256)); }

static void gemm_nt(hipStream_t s, const half_t* A, const half_t* B, float* C, half_t* Ch,
                    const float* bias, int M, int N, int K, int lda, int ldb, int ldc,
                    long long sA, long long sB, long long sC, int batch,
                    float alpha, int relu) {
    dim3 g((N + BN - 1) / BN, (M + BM - 1) / BM, batch);
    gemm16<0><<<g, 256, 0, s>>>(A, B, C, Ch, bias, M, N, K, lda, ldb, ldc, sA, sB, sC, alpha, relu);
}
static void gemm_nn(hipStream_t s, const half_t* A, const half_t* B, float* C, half_t* Ch,
                    const float* bias, int M, int N, int K, int lda, int ldb, int ldc,
                    long long sA, long long sB, long long sC, int batch,
                    float alpha, int relu) {
    dim3 g((N + BN - 1) / BN, (M + BM - 1) / BM, batch);
    gemm16<1><<<g, 256, 0, s>>>(A, B, C, Ch, bias, M, N, K, lda, ldb, ldc, sA, sB, sC, alpha, relu);
}

struct AttnScratch {
    half_t *WQKVh, *WOUTh, *XPADh, *Qh, *Kh, *Vh, *QLh, *KLh;
    half_t *A1h, *A2h, *A3h, *ZH0, *ZH1, *XZh, *Wtmph, *T1h, *T2h, *OUTMh;
    float  *QKVf, *Vf, *SIMf, *SIM2f, *TMPSQf, *OUTf, *PROJf, *scal;
};

static void run_attn(hipStream_t s, float* h,
                     const float* ln_g, const float* ln_b,
                     const float* qkv_w, const float* out_w, const float* out_b,
                     const float* res_w, const AttnScratch& w) {
    const long long SQ = (long long)LAND * LAND;          // 65536
    const long long SV = (long long)NTOK * DH;            // 540672
    const long long SS = (long long)NTOK * LAND;          // 2162688
    const long long SL = (long long)LAND * DH;            // 16384

    k_cast_f16<<<ew_grid((size_t)3 * DIM * DIM), 256, 0, s>>>(qkv_w, w.WQKVh, (size_t)3 * DIM * DIM);
    k_cast_f16<<<ew_grid((size_t)DIM * DIM), 256, 0, s>>>(out_w, w.WOUTh, (size_t)DIM * DIM);

    // LN into front-padded f16 buffer (pad rows zeroed)
    k_zero_f16<<<ew_grid((size_t)BATCH * NTOK * DIM), 256, 0, s>>>(w.XPADh, (size_t)BATCH * NTOK * DIM);
    k_ln_pad<<<BATCH * NSEQ, 256, 0, s>>>(h, ln_g, ln_b, w.XPADh);

    // qkv projection (M = B*NTOK)
    gemm_nt(s, w.XPADh, w.WQKVh, w.QKVf, nullptr, nullptr,
            BATCH * NTOK, 3 * DIM, DIM, DIM, DIM, 3 * DIM, 0, 0, 0, 1, 1.f, 0);
    k_split_qkv<<<ew_grid((size_t)BH * SV), 256, 0, s>>>(w.QKVf, w.Qh, w.Kh, w.Vh, w.Vf);
    k_landmarks<<<ew_grid((size_t)BH * SL), 256, 0, s>>>(w.Qh, w.QLh);
    k_landmarks<<<ew_grid((size_t)BH * SL), 256, 0, s>>>(w.Kh, w.KLh);

    // sim1 = q @ k_l^T  -> softmax -> a1
    gemm_nt(s, w.Qh, w.KLh, w.SIMf, nullptr, nullptr,
            NTOK, LAND, DH, DH, DH, LAND, SV, SL, SS, BH, 1.f, 0);
    k_softmax<<<BH * NTOK, 256, 0, s>>>(w.SIMf, w.A1h, LAND);

    // sim2 = q_l @ k_l^T -> a2
    gemm_nt(s, w.QLh, w.KLh, w.SIM2f, nullptr, nullptr,
            LAND, LAND, DH, DH, DH, LAND, SL, SL, SQ, BH, 1.f, 0);
    k_softmax<<<BH * LAND, 256, 0, s>>>(w.SIM2f, w.A2h, LAND);

    // sim3 = q_l @ k^T -> a3
    gemm_nt(s, w.QLh, w.Kh, w.SIMf, nullptr, nullptr,
            LAND, NTOK, DH, DH, DH, NTOK, SL, SV, SS, BH, 1.f, 0);
    k_softmax<<<BH * LAND, 256, 0, s>>>(w.SIMf, w.A3h, NTOK);

    // Moore-Penrose pinv of a2 (6 Newton iterations)
    k_zero_f32<<<1, 256, 0, s>>>(w.scal, 2);
    k_a2_rowmax<<<(BH * LAND + 255) / 256, 256, 0, s>>>(w.A2h, w.scal);
    k_a2_colmax<<<(BH * LAND + 255) / 256, 256, 0, s>>>(w.A2h, w.scal);
    k_pinv_seed<<<ew_grid((size_t)BH * SQ), 256, 0, s>>>(w.A2h, w.ZH0, w.scal);
    half_t* Zc = w.ZH0;
    half_t* Zn = w.ZH1;
    for (int it = 0; it < 6; ++it) {
        gemm_nn(s, w.A2h, Zc, w.TMPSQf, w.XZh, nullptr,
                LAND, LAND, LAND, LAND, LAND, LAND, SQ, SQ, SQ, BH, 1.f, 0);
        k_ci_minus<<<ew_grid((size_t)BH * SQ), 256, 0, s>>>(7.f, w.TMPSQf, w.Wtmph);
        gemm_nn(s, w.XZh, w.Wtmph, w.TMPSQf, nullptr, nullptr,
                LAND, LAND, LAND, LAND, LAND, LAND, SQ, SQ, SQ, BH, 1.f, 0);
        k_ci_minus<<<ew_grid((size_t)BH * SQ), 256, 0, s>>>(15.f, w.TMPSQf, w.Wtmph);
        gemm_nn(s, w.XZh, w.Wtmph, w.TMPSQf, nullptr, nullptr,
                LAND, LAND, LAND, LAND, LAND, LAND, SQ, SQ, SQ, BH, 1.f, 0);
        k_ci_minus<<<ew_grid((size_t)BH * SQ), 256, 0, s>>>(13.f, w.TMPSQf, w.Wtmph);
        gemm_nn(s, Zc, w.Wtmph, nullptr, Zn, nullptr,
                LAND, LAND, LAND, LAND, LAND, LAND, SQ, SQ, SQ, BH, 0.25f, 0);
        half_t* t = Zc; Zc = Zn; Zn = t;
    }

    // out = a1 @ (Z @ (a3 @ v)) ; + depthwise conv residual
    gemm_nn(s, w.A3h, w.Vh, nullptr, w.T1h, nullptr,
            LAND, DH, NTOK, NTOK, DH, DH, SS, SV, SL, BH, 1.f, 0);
    gemm_nn(s, Zc, w.T1h, nullptr, w.T2h, nullptr,
            LAND, DH, LAND, LAND, DH, DH, SQ, SL, SL, BH, 1.f, 0);
    gemm_nn(s, w.A1h, w.T2h, w.OUTf, nullptr, nullptr,
            NTOK, DH, LAND, LAND, DH, DH, SS, SL, SV, BH, 1.f, 0);
    k_conv_res_add<<<ew_grid((size_t)BH * SV), 256, 0, s>>>(w.Vf, res_w, w.OUTf);

    // merge heads + output projection + residual into h
    k_merge_heads<<<ew_grid((size_t)BATCH * NTOK * DIM), 256, 0, s>>>(w.OUTf, w.OUTMh);
    gemm_nt(s, w.OUTMh, w.WOUTh, w.PROJf, nullptr, out_b,
            BATCH * NTOK, DIM, DIM, DIM, DIM, DIM, 0, 0, 0, 1, 1.f, 0);
    k_add_res<<<ew_grid((size_t)BATCH * NSEQ * DIM), 256, 0, s>>>(h, w.PROJf);
}

// ---------------- entry point ----------------
extern "C" void kernel_launch(void* const* d_in, const int* in_sizes, int n_in,
                              void* d_out, int out_size, void* d_ws, size_t ws_size,
                              hipStream_t stream) {
    (void)in_sizes; (void)n_in; (void)out_size; (void)ws_size;
    const float* data   = (const float*)d_in[0];
    const float* fc1_w  = (const float*)d_in[1];
    const float* fc1_b  = (const float*)d_in[2];
    const float* cls    = (const float*)d_in[3];
    const float* ln1_g  = (const float*)d_in[4];
    const float* ln1_b  = (const float*)d_in[5];
    const float* qkv1_w = (const float*)d_in[6];
    const float* out1_w = (const float*)d_in[7];
    const float* out1_b = (const float*)d_in[8];
    const float* res1_w = (const float*)d_in[9];
    const float* pw7    = (const float*)d_in[10];
    const float* pb7    = (const float*)d_in[11];
    const float* pw5    = (const float*)d_in[12];
    const float* pb5    = (const float*)d_in[13];
    const float* pw3    = (const float*)d_in[14];
    const float* pb3    = (const float*)d_in[15];
    const float* ln2_g  = (const float*)d_in[16];
    const float* ln2_b  = (const float*)d_in[17];
    const float* qkv2_w = (const float*)d_in[18];
    const float* out2_w = (const float*)d_in[19];
    const float* out2_b = (const float*)d_in[20];
    const float* res2_w = (const float*)d_in[21];
    const float* lnf_g  = (const float*)d_in[22];
    const float* lnf_b  = (const float*)d_in[23];
    const float* fc2_w  = (const float*)d_in[24];
    const float* fc2_b  = (const float*)d_in[25];
    float* out = (float*)d_out;

    // ---- workspace bump allocator ----
    size_t off = 0;
    auto alloc = [&](size_t nbytes) -> void* {
        void* p = (char*)d_ws + off;
        off += (nbytes + 255) & ~(size_t)255;
        return p;
    };
    half_t* Xh  = (half_t*)alloc((size_t)BATCH * NIN * OUTER * 2);
    half_t* W1h = (half_t*)alloc((size_t)DIM * OUTER * 2);
    float*  H0f = (float*) alloc((size_t)BATCH * NIN * DIM * 4);
    float*  hA  = (float*) alloc((size_t)BATCH * NSEQ * DIM * 4);
    float*  hB  = (float*) alloc((size_t)BATCH * NSEQ * DIM * 4);

    AttnScratch w;
    w.WQKVh  = (half_t*)alloc((size_t)3 * DIM * DIM * 2);
    w.WOUTh  = (half_t*)alloc((size_t)DIM * DIM * 2);
    w.XPADh  = (half_t*)alloc((size_t)BATCH * NTOK * DIM * 2);
    w.QKVf   = (float*) alloc((size_t)BATCH * NTOK * 3 * DIM * 4);
    w.Qh     = (half_t*)alloc((size_t)BH * NTOK * DH * 2);
    w.Kh     = (half_t*)alloc((size_t)BH * NTOK * DH * 2);
    w.Vh     = (half_t*)alloc((size_t)BH * NTOK * DH * 2);
    w.Vf     = (float*) alloc((size_t)BH * NTOK * DH * 4);
    w.QLh    = (half_t*)alloc((size_t)BH * LAND * DH * 2);
    w.KLh    = (half_t*)alloc((size_t)BH * LAND * DH * 2);
    w.SIMf   = (float*) alloc((size_t)BH * NTOK * LAND * 4);
    w.SIM2f  = (float*) alloc((size_t)BH * LAND * LAND * 4);
    w.A1h    = (half_t*)alloc((size_t)BH * NTOK * LAND * 2);
    w.A2h    = (half_t*)alloc((size_t)BH * LAND * LAND * 2);
    w.A3h    = (half_t*)alloc((size_t)BH * LAND * NTOK * 2);
    w.ZH0    = (half_t*)alloc((size_t)BH * LAND * LAND * 2);
    w.ZH1    = (half_t*)alloc((size_t)BH * LAND * LAND * 2);
    w.XZh    = (half_t*)alloc((size_t)BH * LAND * LAND * 2);
    w.Wtmph  = (half_t*)alloc((size_t)BH * LAND * LAND * 2);
    w.TMPSQf = (float*) alloc((size_t)BH * LAND * LAND * 4);
    w.T1h    = (half_t*)alloc((size_t)BH * LAND * DH * 2);
    w.T2h    = (half_t*)alloc((size_t)BH * LAND * DH * 2);
    w.OUTf   = (float*) alloc((size_t)BH * NTOK * DH * 4);
    w.OUTMh  = (half_t*)alloc((size_t)BATCH * NTOK * DIM * 2);
    w.PROJf  = (float*) alloc((size_t)BATCH * NTOK * DIM * 4);
    w.scal   = (float*) alloc(8 * 4);

    // ---- fc1 + relu ----
    k_cast_f16<<<ew_grid((size_t)BATCH * NIN * OUTER), 256, 0, stream>>>(
        data, Xh, (size_t)BATCH * NIN * OUTER);
    k_cast_f16<<<ew_grid((size_t)DIM * OUTER), 256, 0, stream>>>(
        fc1_w, W1h, (size_t)DIM * OUTER);
    gemm_nt(stream, Xh, W1h, H0f, nullptr, fc1_b,
            BATCH * NIN, DIM, OUTER, OUTER, OUTER, DIM, 0, 0, 0, 1, 1.f, 1);

    // ---- [cls ; h ; wrap] ----
    k_build_h<<<ew_grid((size_t)BATCH * NSEQ * DIM), 256, 0, stream>>>(H0f, cls, hA);

    // ---- block 1 ----
    run_attn(stream, hA, ln1_g, ln1_b, qkv1_w, out1_w, out1_b, res1_w, w);

    // ---- PPEG: hA -> hB ----
    k_copy_cls<<<(BATCH * DIM + 255) / 256, 256, 0, stream>>>(hA, hB);
    k_ppeg<<<ew_grid((size_t)BATCH * HS * HS * DIM), 256, 0, stream>>>(
        hA, hB, pw7, pb7, pw5, pb5, pw3, pb3);

    // ---- block 2 ----
    run_attn(stream, hB, ln2_g, ln2_b, qkv2_w, out2_w, out2_b, res2_w, w);

    // ---- final LN + head ----
    k_final<<<BATCH, 256, 0, stream>>>(hB, lnf_g, lnf_b, fc2_w, fc2_b, out);
}
